// WMSA_4715874091086
// MI455X (gfx1250) — matrix operand order repaired and checked by reading, not compiled, not run on hardware
//
#include <hip/hip_runtime.h>
#include <hip/hip_bf16.h>

// ---------------------------------------------------------------------------
// Swin W-MSA (shifted window attention), fully fused per-window kernel.
// B=16, H=W=64, C=512, p=8 -> 1024 windows of 64 tokens; 16 heads, hd=32.
// All GEMMs run on v_wmma_f32_16x16x32_bf16 (fp32 accumulate).
// Weights are pre-packed into a WMMA-B-fragment-native layout so every lane
// loads its 16 B elements as two global_load_b128 (no per-element packing).
// ---------------------------------------------------------------------------

typedef __attribute__((ext_vector_type(16))) __bf16 v16bf;
typedef __attribute__((ext_vector_type(8)))  float  v8f;

#define N_TOK    64          // tokens per window (8x8)
#define CCH      512         // channels
#define HEADS    16
#define HD       32
#define KC512    32          // number of 16-wide K chunks in K=512
#define LDA      520         // bf16 ld for xw / out buffer (pad vs 512)
#define LDQ      36          // bf16 ld for per-head Q/K tiles (pad vs 32)
#define LDVT     66          // bf16 ld for transposed V (32 x 64, pad vs 64)
#define LDP      68          // bf16 ld for P tile (pad vs 64)
#define SCALE_Q  0.17677669529663687f   // 1/sqrt(32)

#define XW_BYTES   (N_TOK * LDA * 2)                    // 66560
#define REL_OFF    XW_BYTES
#define REL_BYTES  (225 * HEADS * 4)                    // 14400
#define PW_OFF     (REL_OFF + REL_BYTES)                // 80960
#define PERWAVE    (3 * N_TOK * LDQ * 2)                // 13824 (Q,K,V slots)
#define LDS_TOTAL  (PW_OFF + HEADS * PERWAVE)           // 302144 bytes

// ----- WMMA fragment loaders (CDNA5 wave32 VGPR layouts) --------------------

// A-matrix 16x32 bf16 tile from row-major bf16 buffer (row = M, col = K).
// lane: M = lane&15 ; lane-half selects K in {16*kh..16*kh+15} (contiguous).
__device__ __forceinline__ v16bf ldsA(const __bf16* base, int row0, int ld, int k0) {
  const int l  = threadIdx.x & 31;
  const __bf16* p = base + (row0 + (l & 15)) * ld + k0 + ((l >> 4) << 4);
  v16bf a;
#pragma unroll
  for (int e = 0; e < 8; ++e) { a[e] = p[e]; a[8 + e] = p[8 + e]; }
  return a;
}

// B-matrix 32x16 tile where element (k,n) is stored at base[n*row_ld + k]
// (i.e. the source is laid out N-major with contiguous K). Lane reads its 16
// K-values contiguously -> two ds_load_b128.
__device__ __forceinline__ v16bf ldsBrow(const __bf16* base, int row_ld, int n0, int k0) {
  const int l  = threadIdx.x & 31;
  const __bf16* p = base + (n0 + (l & 15)) * row_ld + k0 + ((l >> 4) << 4);
  v16bf b;
#pragma unroll
  for (int e = 0; e < 16; ++e) b[e] = p[e];
  return b;
}

// B-matrix 32x16 tile from the packed global weight image:
//   pk[((nb*KC + kc)*16 + n)*16 + e] = W[(kc*16+e)*N + nb*16 + n]
// Lane's 16 elements are contiguous and 32B-aligned -> two global_load_b128.
__device__ __forceinline__ v16bf glbBpk(const __bf16* base, int KC, int kt, int n0) {
  const int l  = threadIdx.x & 31;
  const int kc = kt * 2 + (l >> 4);
  const size_t off = ((((size_t)(n0 >> 4) * KC + kc) * 16 + (l & 15)) << 4);
  return *(const v16bf*)(base + off);
}

__device__ __forceinline__ v8f wmma_bf16(v16bf a, v16bf b, v8f c) {
  return __builtin_amdgcn_wmma_f32_16x16x32_bf16(
      /*neg_a=*/false, a, /*neg_b=*/false, b,
      /*c_mod=*/(short)0, c, /*reuse_a=*/false, /*reuse_b=*/false);
}

// 64x32 = (64x512 LDS bf16) @ (32-col slice of packed global bf16 W) + bias,
// *scale, stored as bf16 to LDS; optionally transposed (for V).
template <bool TRANSPOSE_DST>
__device__ __forceinline__ void gemm_qkv(const __bf16* A, const __bf16* Wpk,
                                         int col0, const float* bias,
                                         float scale, __bf16* dst, int ldd) {
  v8f acc[4][2] = {};
  for (int kt = 0; kt < 16; ++kt) {
    const v16bf b0 = glbBpk(Wpk, KC512, kt, col0);
    const v16bf b1 = glbBpk(Wpk, KC512, kt, col0 + 16);
#pragma unroll
    for (int mt = 0; mt < 4; ++mt) {
      const v16bf a = ldsA(A, mt * 16, LDA, kt * 32);
      acc[mt][0] = wmma_bf16(a, b0, acc[mt][0]);
      acc[mt][1] = wmma_bf16(a, b1, acc[mt][1]);
    }
  }
  const int l = threadIdx.x & 31;
  const int nlo = l & 15, mh = (l >> 4) << 3;
#pragma unroll
  for (int nt = 0; nt < 2; ++nt) {
    const float bv = bias[col0 + nt * 16 + nlo];
#pragma unroll
    for (int mt = 0; mt < 4; ++mt) {
      if (TRANSPOSE_DST) {
        // dst[d][token]: contiguous 8-token run -> ds_store_b128
        __bf16* p = dst + (nt * 16 + nlo) * ldd + mt * 16 + mh;
#pragma unroll
        for (int r = 0; r < 8; ++r)
          p[r] = (__bf16)((acc[mt][nt][r] + bv) * scale);
      } else {
        __bf16* p = dst + (mt * 16 + mh) * ldd + nt * 16 + nlo;
#pragma unroll
        for (int r = 0; r < 8; ++r)
          p[r * ldd] = (__bf16)((acc[mt][nt][r] + bv) * scale);
      }
    }
  }
}

// ----- one-time weight fp32 -> packed bf16 conversion -----------------------
__global__ void wmsa_pack_weights(const float* __restrict__ wqkv,
                                  const float* __restrict__ wout,
                                  __bf16* __restrict__ dq,
                                  __bf16* __restrict__ dw) {
  const int i = blockIdx.x * blockDim.x + threadIdx.x;
  if (i < CCH * 3 * CCH) {
    const int e = i & 15, n = (i >> 4) & 15;
    const int kc = (i >> 8) & 31, nb = i >> 13;   // KC=32
    dq[i] = (__bf16)wqkv[(kc * 16 + e) * (3 * CCH) + nb * 16 + n];
  }
  if (i < CCH * CCH) {
    const int e = i & 15, n = (i >> 4) & 15;
    const int kc = (i >> 8) & 31, nb = i >> 13;
    dw[i] = (__bf16)wout[(kc * 16 + e) * CCH + nb * 16 + n];
  }
}

// ----- fused per-window kernel ----------------------------------------------
__global__ void __launch_bounds__(512, 1)
wmsa_fused(const float* __restrict__ x, const float* __restrict__ bqkv,
           const float* __restrict__ rel, const float* __restrict__ bout,
           const __bf16* __restrict__ wqkvB, const __bf16* __restrict__ woutB,
           float* __restrict__ out) {
  extern __shared__ char smem[];
  __bf16* xw   = (__bf16*)smem;                 // 64 x 512 window (later: out)
  float*  relS = (float*)(smem + REL_OFF);      // 225 x 16 rel-pos table
  const int tid = threadIdx.x;
  const int h   = tid >> 5;                     // wave id == head id
  const int l   = tid & 31;
  const int nlo = l & 15, mh = (l >> 4) << 3;

  __bf16* Qb = (__bf16*)(smem + PW_OFF + h * PERWAVE);
  __bf16* Kb = Qb + N_TOK * LDQ;
  __bf16* Vt = Kb + N_TOK * LDQ;                // transposed V: 32 x 64 (LDVT)
  __bf16* Pb = Qb;                              // P overlays dead Q+K region

  const int bw = blockIdx.x;                    // window id
  const int b  = bw >> 6, wy = (bw >> 3) & 7, wx = bw & 7;

  // ---- Phase 0: gather rolled window (bf16) + rel_pos into LDS -------------
  {
    const int row = tid >> 3;                   // 64 rows, 8 threads per row
    const int c0  = (tid & 7) * 64;
    const int py = row >> 3, px = row & 7;
    const int gy = ((wy << 3) + py + 4) & 63;   // roll(-shift) + window
    const int gx = ((wx << 3) + px + 4) & 63;
    const float4* src =
        (const float4*)(x + ((((size_t)b << 6) + gy) * 64 + gx) * CCH + c0);
    __bf16* dst = xw + row * LDA + c0;
#pragma unroll
    for (int i = 0; i < 16; ++i) {
      const float4 v = src[i];
      dst[i * 4 + 0] = (__bf16)v.x; dst[i * 4 + 1] = (__bf16)v.y;
      dst[i * 4 + 2] = (__bf16)v.z; dst[i * 4 + 3] = (__bf16)v.w;
    }
  }
  for (int i = tid; i < 225 * HEADS; i += 512) relS[i] = rel[i];
  if (l == 0) {  // warm L2 with this wave's weight slices
    __builtin_prefetch(wqkvB + ((size_t)(h * 2) * KC512) * 256, 0, 2);
    __builtin_prefetch(woutB + ((size_t)(h * 2) * KC512) * 256, 0, 2);
  }
  __syncthreads();

  // ---- Phase 1: per-head QKV projection (3 x 64x32 GEMM, K=512) ------------
  gemm_qkv<false>(xw, wqkvB, h * HD,            bqkv, SCALE_Q, Qb, LDQ);
  gemm_qkv<false>(xw, wqkvB, CCH + h * HD,      bqkv, 1.0f,    Kb, LDQ);
  gemm_qkv<true >(xw, wqkvB, 2 * CCH + h * HD,  bqkv, 1.0f,    Vt, LDVT);
  __syncthreads();   // all xw reads done; xw becomes the attention-out buffer

  // ---- Phase 2: S = Q K^T + bias (registers) -------------------------------
  v8f s[4][4];
  {
#pragma unroll
    for (int nt = 0; nt < 4; ++nt) {
      const v16bf kb = ldsBrow(Kb, LDQ, nt * 16, 0);   // K^T fragment
#pragma unroll
      for (int mt = 0; mt < 4; ++mt) {
        const v16bf q = ldsA(Qb, mt * 16, LDQ, 0);
        const v8f z = {};
        s[mt][nt] = wmma_bf16(q, kb, z);
      }
    }
  }
  // relative-position bias
#pragma unroll
  for (int nt = 0; nt < 4; ++nt) {
    const int n = nt * 16 + nlo, ky = n >> 3, kx = n & 7;
#pragma unroll
    for (int mt = 0; mt < 4; ++mt)
#pragma unroll
      for (int r = 0; r < 8; ++r) {
        const int m = mt * 16 + mh + r, qy = m >> 3, qx = m & 7;
        s[mt][nt][r] += relS[((qy - ky + 7) * 15 + (qx - kx + 7)) * HEADS + h];
      }
  }
  // ---- Phase 3: row softmax (rows live inside one 16-lane group) -----------
#pragma unroll
  for (int mt = 0; mt < 4; ++mt)
#pragma unroll
    for (int r = 0; r < 8; ++r) {
      float mx = fmaxf(fmaxf(s[mt][0][r], s[mt][1][r]),
                       fmaxf(s[mt][2][r], s[mt][3][r]));
      mx = fmaxf(mx, __shfl_xor(mx, 1, 32));
      mx = fmaxf(mx, __shfl_xor(mx, 2, 32));
      mx = fmaxf(mx, __shfl_xor(mx, 4, 32));
      mx = fmaxf(mx, __shfl_xor(mx, 8, 32));
      float sum = 0.f;
#pragma unroll
      for (int nt = 0; nt < 4; ++nt) {
        const float e = __expf(s[mt][nt][r] - mx);
        s[mt][nt][r] = e; sum += e;
      }
      sum += __shfl_xor(sum, 1, 32);
      sum += __shfl_xor(sum, 2, 32);
      sum += __shfl_xor(sum, 4, 32);
      sum += __shfl_xor(sum, 8, 32);
      const float inv = 1.f / sum;
#pragma unroll
      for (int nt = 0; nt < 4; ++nt) s[mt][nt][r] *= inv;
    }
  // P -> bf16 in LDS (overwrites Q/K; same-wave ordering via DScnt)
#pragma unroll
  for (int nt = 0; nt < 4; ++nt)
#pragma unroll
    for (int mt = 0; mt < 4; ++mt) {
      __bf16* p = Pb + (mt * 16 + mh) * LDP + nt * 16 + nlo;
#pragma unroll
      for (int r = 0; r < 8; ++r) p[r * LDP] = (__bf16)s[mt][nt][r];
    }

  // ---- Phase 4: O = P @ V  (V^T is N-major -> contiguous B loads) ----------
  v8f o[4][2] = {};
#pragma unroll
  for (int kt = 0; kt < 2; ++kt) {
    const v16bf bv0 = ldsBrow(Vt, LDVT, 0,  kt * 32);
    const v16bf bv1 = ldsBrow(Vt, LDVT, 16, kt * 32);
#pragma unroll
    for (int mt = 0; mt < 4; ++mt) {
      const v16bf a = ldsA(Pb, mt * 16, LDP, kt * 32);
      o[mt][0] = wmma_bf16(a, bv0, o[mt][0]);
      o[mt][1] = wmma_bf16(a, bv1, o[mt][1]);
    }
  }
  // store head output (bf16) into the shared 64x512 out buffer (reused xw)
#pragma unroll
  for (int nt = 0; nt < 2; ++nt)
#pragma unroll
    for (int mt = 0; mt < 4; ++mt) {
      __bf16* p = xw + (mt * 16 + mh) * LDA + h * HD + nt * 16 + nlo;
#pragma unroll
      for (int r = 0; r < 8; ++r) p[r * LDA] = (__bf16)o[mt][nt][r];
    }
  __syncthreads();

  // ---- Phase 5: out projection, wave h owns output cols [h*32, h*32+32) ----
  v8f acc[4][2] = {};
  for (int kt = 0; kt < 16; ++kt) {
    const v16bf b0 = glbBpk(woutB, KC512, kt, h * HD);
    const v16bf b1 = glbBpk(woutB, KC512, kt, h * HD + 16);
#pragma unroll
    for (int mt = 0; mt < 4; ++mt) {
      const v16bf a = ldsA(xw, mt * 16, LDA, kt * 32);
      acc[mt][0] = wmma_bf16(a, b0, acc[mt][0]);
      acc[mt][1] = wmma_bf16(a, b1, acc[mt][1]);
    }
  }
  // epilogue: + b_out, inverse roll/window scatter to global fp32
#pragma unroll
  for (int nt = 0; nt < 2; ++nt) {
    const int c = h * HD + nt * 16 + nlo;
    const float bo = bout[c];
#pragma unroll
    for (int mt = 0; mt < 4; ++mt)
#pragma unroll
      for (int r = 0; r < 8; ++r) {
        const int m  = mt * 16 + mh + r;
        const int py = m >> 3, px = m & 7;
        const int gy = ((wy << 3) + py + 4) & 63;
        const int gx = ((wx << 3) + px + 4) & 63;
        out[((((size_t)b << 6) + gy) * 64 + gx) * CCH + c] = acc[mt][nt][r] + bo;
      }
  }
}

// ---------------------------------------------------------------------------
extern "C" void kernel_launch(void* const* d_in, const int* in_sizes, int n_in,
                              void* d_out, int out_size, void* d_ws, size_t ws_size,
                              hipStream_t stream) {
  const float* x    = (const float*)d_in[0];
  const float* wqkv = (const float*)d_in[1];
  const float* bqkv = (const float*)d_in[2];
  const float* rel  = (const float*)d_in[3];
  const float* wout = (const float*)d_in[4];
  const float* bout = (const float*)d_in[5];
  // d_in[6] = window_size (always 8; hardcoded)
  float* out = (float*)d_out;

  __bf16* wsQ = (__bf16*)d_ws;                 // packed 512*1536 bf16
  __bf16* wsO = wsQ + (size_t)CCH * 3 * CCH;   // packed 512*512  bf16

  wmsa_pack_weights<<<(CCH * 3 * CCH + 255) / 256, 256, 0, stream>>>(
      wqkv, wout, wsQ, wsO);

  (void)hipFuncSetAttribute((const void*)wmsa_fused,
                            hipFuncAttributeMaxDynamicSharedMemorySize,
                            (int)LDS_TOTAL);
  wmsa_fused<<<1024, 512, LDS_TOTAL, stream>>>(x, bqkv, rel, bout, wsQ, wsO, out);
}
